// MultiHeadAttention_7464653160853
// MI455X (gfx1250) — compile-verified, matrix operand order checked by reading, not run on hardware
//
#include <hip/hip_runtime.h>
#include <math.h>

// ---------------------------------------------------------------------------
// MultiHeadAttention for MI455X (gfx1250): B=8, S=2048, D=512, H=8, hd=64
// fp32 inputs -> f16 staged tensors -> v_wmma_f32_16x16x32_f16 everywhere,
// flash-attention streaming softmax (never materializes S x S scores).
// ---------------------------------------------------------------------------

typedef _Float16 v16h __attribute__((ext_vector_type(16)));
typedef _Float16 v8h  __attribute__((ext_vector_type(8)));
typedef float    v8f  __attribute__((ext_vector_type(8)));

constexpr int NB = 8;     // batch
constexpr int NS = 2048;  // sequence
constexpr int ND = 512;   // d_model == d_embed
constexpr int NH = 8;     // heads
constexpr int HD = 64;    // head dim

// CDNA5 16-bit A/B fragment layout: lane = (m = lane&15, half = lane>>4),
// element i holds K = (i<8 ? i : i+8) + 8*half  => two contiguous runs of 8
// halfs at byte offsets {16*half, 32 + 16*half}. So a fragment is two 16-byte
// loads from (base + 8*half) and (base + 8*half + 16 halfs).
__device__ __forceinline__ v16h load_frag_h16(const _Float16* p) {
  v8h lo = *(const v8h*)(p);
  v8h hi = *(const v8h*)(p + 16);
  v16h r;
#pragma unroll
  for (int i = 0; i < 8; ++i) { r[i] = lo[i]; r[i + 8] = hi[i]; }
  return r;
}

__device__ __forceinline__ v16h load_frag_f32_cvt(const float* p) {
  v16h r;
#pragma unroll
  for (int i = 0; i < 8; ++i) {
    r[i]     = (_Float16)p[i];
    r[i + 8] = (_Float16)p[16 + i];
  }
  return r;
}

__device__ __forceinline__ v8f wmma16(v16h a, v16h b, v8f c) {
  // D(16x16 f32) = A(16x32 f16) * B(32x16 f16) + C
  return __builtin_amdgcn_wmma_f32_16x16x32_f16(
      false, a, false, b, (short)0, c, false, false);
}

// ---------------------------------------------------------------------------
// Pre-transpose + fp32->f16 convert the 4 weight matrices: WT[mat][n][k].
// ---------------------------------------------------------------------------
__global__ void __launch_bounds__(256) prep_weights_kernel(
    const float* __restrict__ Wq, const float* __restrict__ Wk,
    const float* __restrict__ Wv, const float* __restrict__ Wo,
    _Float16* __restrict__ WT) {
  int idx = blockIdx.x * 256 + threadIdx.x;  // 4 * 512 * 512 total
  int mat = idx >> 18;
  int rem = idx & ((1 << 18) - 1);
  int k = rem >> 9;
  int n = rem & 511;
  const float* W = (mat == 0) ? Wq : (mat == 1) ? Wk : (mat == 2) ? Wv : Wo;
  WT[((size_t)mat << 18) + (size_t)n * ND + k] = (_Float16)W[(size_t)k * ND + n];
}

// ---------------------------------------------------------------------------
// Projection GEMM: Y = X(fp32 [M=B*S, 512]) @ W + bias, stored f16 head-split.
//   mode 0/1 : out[b][h][s][d]   (Q, K)
//   mode 2   : out[b][h][d][s]   (V transposed so P@V B-fragments are contiguous)
// One wave computes a 16x64 output tile (4 WMMA accumulators, A reused 4x).
// ---------------------------------------------------------------------------
__global__ void __launch_bounds__(128) proj_qkv_kernel(
    const float* __restrict__ X, const _Float16* __restrict__ WT,
    const float* __restrict__ bias, _Float16* __restrict__ out, int mode) {
  const int lane = threadIdx.x & 31;
  const int wave = threadIdx.x >> 5;
  const int gw = blockIdx.x * 4 + wave;  // (M/16)*(N/64) = 1024*8 = 8192 waves
  const int rt = gw >> 3;                // row tile
  const int ng = gw & 7;                 // 64-wide column group
  const int m = lane & 15, half = lane >> 4;
  const int row = rt * 16 + m;

  v8f vzero = {};
  v8f acc[4];
#pragma unroll
  for (int nt = 0; nt < 4; ++nt) acc[nt] = vzero;

  for (int kk = 0; kk < ND; kk += 32) {
    const float* xp = X + (size_t)row * ND + kk + 8 * half;
    v16h a = load_frag_f32_cvt(xp);
#pragma unroll
    for (int nt = 0; nt < 4; ++nt) {
      int n = ng * 64 + nt * 16 + m;
      v16h b = load_frag_h16(WT + (size_t)n * ND + kk + 8 * half);
      acc[nt] = wmma16(a, b, acc[nt]);
    }
  }

#pragma unroll
  for (int nt = 0; nt < 4; ++nt) {
    int col = ng * 64 + nt * 16 + m;
    int h_ = col >> 6, d_ = col & 63;
    float bvv = bias[col];
#pragma unroll
    for (int r = 0; r < 8; ++r) {
      int rowg = rt * 16 + r + 8 * half;  // C-layout: VGPR r holds M = r + 8*half
      int b_ = rowg >> 11, s_ = rowg & (NS - 1);
      float v = acc[nt][r] + bvv;
      size_t idx = (mode < 2)
          ? ((((size_t)b_ * NH + h_) * NS + s_) * HD + d_)
          : ((((size_t)b_ * NH + h_) * HD + d_) * NS + s_);
      out[idx] = (_Float16)v;
    }
  }
}

// ---------------------------------------------------------------------------
// Flash attention: one wave per (b,h, 16-row Q tile). Streams 32-key chunks:
//   4 WMMAs for Q*K^T (d=64 = 2 chained K=32 contractions x 2 key sub-tiles),
//   online softmax with half-wave (16-lane) row reductions,
//   P relayout via per-wave LDS, 4 WMMAs for P@V.
// ---------------------------------------------------------------------------
__global__ void __launch_bounds__(128) attn_kernel(
    const _Float16* __restrict__ Qh, const _Float16* __restrict__ Kh,
    const _Float16* __restrict__ Vt, _Float16* __restrict__ Ctx) {
  __shared__ _Float16 Ps[4][16][32];  // per-wave P staging (C-layout -> A-layout)

  const int lane = threadIdx.x & 31;
  const int wave = threadIdx.x >> 5;
  const int gw = blockIdx.x * 4 + wave;  // B*H*(S/16) = 8192 waves
  const int qt = gw & (NS / 16 - 1);     // q tile (0..127)
  const int bh = gw >> 7;                // fused batch*head
  const int m = lane & 15, half = lane >> 4;
  const float scale = 0.044194173824159216f;  // 1/sqrt(512) (d_embed, per ref)

  const _Float16* Qp = Qh + ((size_t)bh * NS + qt * 16 + m) * HD;
  const _Float16* Kp = Kh + (size_t)bh * NS * HD;
  const _Float16* Vp = Vt + (size_t)bh * HD * NS;

  // Q fragments (row m, d-chunks 0..31 and 32..63), loaded once
  v16h a0 = load_frag_h16(Qp + 8 * half);
  v16h a1 = load_frag_h16(Qp + 32 + 8 * half);

  float mstat[8], lstat[8];
  v8f vzero = {};
  v8f acc[4];
#pragma unroll
  for (int r = 0; r < 8; ++r) { mstat[r] = -1e30f; lstat[r] = 0.0f; }
#pragma unroll
  for (int dt = 0; dt < 4; ++dt) acc[dt] = vzero;

  const int nchunks = qt / 2 + 1;  // causal: last chunk covers the diagonal
  for (int jc = 0; jc < nchunks; ++jc) {
    const int kb = jc * 32;

    // ---- scores: two 16x16 f32 tiles (keys kb..kb+15 and kb+16..kb+31)
    v8f s[2];
#pragma unroll
    for (int t = 0; t < 2; ++t) {
      const _Float16* kp = Kp + (size_t)(kb + t * 16 + m) * HD;  // B col = key
      v16h b0 = load_frag_h16(kp + 8 * half);        // d 0..31
      v16h b1 = load_frag_h16(kp + 32 + 8 * half);   // d 32..63
      s[t] = wmma16(a0, b0, vzero);
      s[t] = wmma16(a1, b1, s[t]);
    }

    // ---- scale + causal mask (only the final chunk straddles the diagonal)
    const bool last = (jc == nchunks - 1);
#pragma unroll
    for (int t = 0; t < 2; ++t) {
#pragma unroll
      for (int r = 0; r < 8; ++r) {
        float v = s[t][r] * scale;
        if (last) {
          int key = kb + t * 16 + m;
          int row = qt * 16 + r + 8 * half;
          if (key > row) v = -1e9f;
        }
        s[t][r] = v;
      }
    }

    // ---- online softmax (row = r + 8*half, columns spread over 16 lanes)
    float p0[8], p1[8];
#pragma unroll
    for (int r = 0; r < 8; ++r) {
      float rmax = fmaxf(s[0][r], s[1][r]);
#pragma unroll
      for (int off = 1; off < 16; off <<= 1)
        rmax = fmaxf(rmax, __shfl_xor(rmax, off));
      float mnew = fmaxf(mstat[r], rmax);
      float corr = __expf(mstat[r] - mnew);
      mstat[r] = mnew;
      p0[r] = __expf(s[0][r] - mnew);
      p1[r] = __expf(s[1][r] - mnew);
      float rs = p0[r] + p1[r];
#pragma unroll
      for (int off = 1; off < 16; off <<= 1)
        rs += __shfl_xor(rs, off);
      lstat[r] = lstat[r] * corr + rs;
#pragma unroll
      for (int dt = 0; dt < 4; ++dt) acc[dt][r] *= corr;
    }

    // ---- P: C-layout -> A-layout via LDS (per-wave region; same-wave DS ops
    //      are architecturally in-order, no barrier needed)
#pragma unroll
    for (int r = 0; r < 8; ++r) {
      Ps[wave][r + 8 * half][m]      = (_Float16)p0[r];
      Ps[wave][r + 8 * half][16 + m] = (_Float16)p1[r];
    }
    const _Float16* pp = &Ps[wave][m][0];
    v16h pa = load_frag_h16(pp + 8 * half);  // 16x32 f16 A-fragment of P

    // ---- ctx += P @ V  (V stored [d][s], so B-fragments are contiguous)
#pragma unroll
    for (int dt = 0; dt < 4; ++dt) {
      const _Float16* vp = Vp + (size_t)(dt * 16 + m) * NS + kb;
      v16h bv = load_frag_h16(vp + 8 * half);
      acc[dt] = wmma16(pa, bv, acc[dt]);
    }
  }

  // ---- normalize by l and store context, layout [b][h][s][d] in f16
#pragma unroll
  for (int r = 0; r < 8; ++r) {
    float inv = 1.0f / lstat[r];
    int row = qt * 16 + r + 8 * half;
    _Float16* op = Ctx + ((size_t)bh * NS + row) * HD;
#pragma unroll
    for (int dt = 0; dt < 4; ++dt)
      op[dt * 16 + m] = (_Float16)(acc[dt][r] * inv);
  }
}

// ---------------------------------------------------------------------------
// Output projection: Y(fp32) = Ctx(f16 [b,h,s,d] viewed as [M,512]) @ Wo + bo
// ---------------------------------------------------------------------------
__global__ void __launch_bounds__(128) proj_out_kernel(
    const _Float16* __restrict__ Ctx, const _Float16* __restrict__ WoT,
    const float* __restrict__ bo, float* __restrict__ Y) {
  const int lane = threadIdx.x & 31;
  const int wave = threadIdx.x >> 5;
  const int gw = blockIdx.x * 4 + wave;
  const int rt = gw >> 3;
  const int ng = gw & 7;
  const int m = lane & 15, half = lane >> 4;
  const int row = rt * 16 + m;
  const int b_ = row >> 11, s_ = row & (NS - 1);

  v8f vzero = {};
  v8f acc[4];
#pragma unroll
  for (int nt = 0; nt < 4; ++nt) acc[nt] = vzero;

  for (int kk = 0; kk < ND; kk += 32) {
    // k = h*64 + d ; a 32-chunk never crosses a head boundary
    int h_ = kk >> 6;
    const _Float16* ap =
        Ctx + (((size_t)b_ * NH + h_) * NS + s_) * HD + (kk & 63) + 8 * half;
    v16h a = load_frag_h16(ap);
#pragma unroll
    for (int nt = 0; nt < 4; ++nt) {
      int n = ng * 64 + nt * 16 + m;
      v16h b = load_frag_h16(WoT + (size_t)n * ND + kk + 8 * half);
      acc[nt] = wmma16(a, b, acc[nt]);
    }
  }

#pragma unroll
  for (int nt = 0; nt < 4; ++nt) {
    int col = ng * 64 + nt * 16 + m;
    float bvv = bo[col];
#pragma unroll
    for (int r = 0; r < 8; ++r) {
      int rowg = rt * 16 + r + 8 * half;
      Y[(size_t)rowg * ND + col] = acc[nt][r] + bvv;
    }
  }
}

// ---------------------------------------------------------------------------
extern "C" void kernel_launch(void* const* d_in, const int* in_sizes, int n_in,
                              void* d_out, int out_size, void* d_ws,
                              size_t ws_size, hipStream_t stream) {
  (void)in_sizes; (void)n_in; (void)out_size; (void)ws_size;
  const float* q  = (const float*)d_in[0];
  const float* k  = (const float*)d_in[1];
  const float* v  = (const float*)d_in[2];
  // d_in[3] = dense causal mask: applied analytically, not read
  const float* Wq = (const float*)d_in[4];
  const float* bq = (const float*)d_in[5];
  const float* Wk = (const float*)d_in[6];
  const float* bk = (const float*)d_in[7];
  const float* Wv = (const float*)d_in[8];
  const float* bv = (const float*)d_in[9];
  const float* Wo = (const float*)d_in[10];
  const float* bo = (const float*)d_in[11];

  const size_t szBH = (size_t)NB * NH * NS * HD;  // 8,388,608 f16 elems
  _Float16* ws  = (_Float16*)d_ws;
  _Float16* Qh  = ws;
  _Float16* Kh  = Qh + szBH;
  _Float16* Vt  = Kh + szBH;
  _Float16* Ctx = Vt + szBH;
  _Float16* WT  = Ctx + szBH;  // 4 * 512*512 f16

  prep_weights_kernel<<<4096, 256, 0, stream>>>(Wq, Wk, Wv, Wo, WT);
  proj_qkv_kernel<<<2048, 128, 0, stream>>>(q, WT,                 bq, Qh, 0);
  proj_qkv_kernel<<<2048, 128, 0, stream>>>(k, WT + (1 << 18),     bk, Kh, 1);
  proj_qkv_kernel<<<2048, 128, 0, stream>>>(v, WT + (2 << 18),     bv, Vt, 2);
  attn_kernel<<<2048, 128, 0, stream>>>(Qh, Kh, Vt, Ctx);
  proj_out_kernel<<<2048, 128, 0, stream>>>(Ctx, WT + ((size_t)3 << 18), bo,
                                            (float*)d_out);
}
